// muPMultiHeadAttention_52879637348685
// MI455X (gfx1250) — matrix operand order repaired and checked by reading, not compiled, not run on hardware
//
#include <hip/hip_runtime.h>
#include <hip/hip_bf16.h>

typedef __attribute__((ext_vector_type(16))) __bf16 v16bf;
typedef __attribute__((ext_vector_type(8)))  __bf16 v8bf;
typedef __attribute__((ext_vector_type(8)))  float  v8f;

#define S_LEN  4096
#define DMODEL 1024
#define NHEAD  16
#define HDIM   64

// D = A(16x32 bf16) * B(32x16 bf16) + C(16x16 f32)
static __device__ inline v8f wmma_bf16(v16bf a, v16bf b, v8f c) {
  return __builtin_amdgcn_wmma_f32_16x16x32_bf16(
      /*neg_a=*/false, a, /*neg_b=*/false, b,
      /*c_mod=*/(short)0, c, /*reuse_a=*/false, /*reuse_b=*/false);
}

// A-fragment (16x32): lane m = lane&15; elem e -> K = (e<8?e:e+8) + (lane>=16?8:0)
static __device__ inline v16bf load_a_bf16(const __bf16* __restrict__ row, int lane) {
  const int koff = (lane & 16) ? 8 : 0;
  const v8bf lo = *reinterpret_cast<const v8bf*>(row + koff);
  const v8bf hi = *reinterpret_cast<const v8bf*>(row + koff + 16);
  v16bf a;
#pragma unroll
  for (int e = 0; e < 8; ++e) { a[e] = lo[e]; a[e + 8] = hi[e]; }
  return a;
}

// B-fragment (32x16): lane n = lane&15; elem e -> K = (e<8?e:e+8) + (lane>=16?8:0)
static __device__ inline v16bf load_b_bf16(const __bf16* __restrict__ row, int lane) {
  const int koff = (lane & 16) ? 8 : 0;
  const v8bf lo = *reinterpret_cast<const v8bf*>(row + koff);
  const v8bf hi = *reinterpret_cast<const v8bf*>(row + koff + 16);
  v16bf b;
#pragma unroll
  for (int e = 0; e < 8; ++e) { b[e] = lo[e]; b[e + 8] = hi[e]; }
  return b;
}

// ---------------- Kernel 0: one-shot f32 -> bf16 conversion ----------------
__global__ __launch_bounds__(256)
void cvt_bf16_kernel(const float* __restrict__ src, __bf16* __restrict__ dst, int n8) {
  const int i = blockIdx.x * 256 + threadIdx.x;
  if (i >= n8) return;
  const float4 f0 = reinterpret_cast<const float4*>(src)[2 * i];
  const float4 f1 = reinterpret_cast<const float4*>(src)[2 * i + 1];
  v8bf o;
  o[0]=(__bf16)f0.x; o[1]=(__bf16)f0.y; o[2]=(__bf16)f0.z; o[3]=(__bf16)f0.w;
  o[4]=(__bf16)f1.x; o[5]=(__bf16)f1.y; o[6]=(__bf16)f1.z; o[7]=(__bf16)f1.w;
  reinterpret_cast<v8bf*>(dst)[i] = o;
}

// ---------------- Kernel A: fused QKV projection + RoPE ----------------
// grid(S/16, H), block 32. Each wave: 16 s-rows x one full head (64 cols)
// of Q, K, V => 12 WMMAs per k-step off a single shared A-fragment.
__global__ __launch_bounds__(32)
void qkv_rope_kernel(const __bf16* __restrict__ xb,
                     const __bf16* __restrict__ wqb,
                     const __bf16* __restrict__ wkb,
                     const __bf16* __restrict__ wvb,
                     __bf16* __restrict__ qws,   // [H][S][HD]
                     __bf16* __restrict__ kws,   // [H][S][HD]
                     __bf16* __restrict__ vtws)  // [H][HD][S] (transposed)
{
  const int lane = threadIdx.x;
  const int st = blockIdx.x;            // 16-row tile along S
  const int h  = blockIdx.y;            // head -> output cols h*64 .. h*64+63
  const int s_a  = st * 16 + (lane & 15);
  const int brow = lane & 15;

  v8f aq[4] = {}; v8f ak[4] = {}; v8f av[4] = {};
  for (int kk = 0; kk < DMODEL; kk += 32) {
    const v16bf a = load_a_bf16(xb + (size_t)s_a * DMODEL + kk, lane);
#pragma unroll
    for (int j = 0; j < 4; ++j) {
      const size_t wrow = (size_t)(h * HDIM + j * 16 + brow) * DMODEL + kk;
      aq[j] = wmma_bf16(a, load_b_bf16(wqb + wrow, lane), aq[j]);
      ak[j] = wmma_bf16(a, load_b_bf16(wkb + wrow, lane), ak[j]);
      av[j] = wmma_bf16(a, load_b_bf16(wvb + wrow, lane), av[j]);
    }
  }

  // C layout: col n = lane&15, rows M = r + (lane>=16 ? 8 : 0)
  const int n = lane & 15;
  const int rbase = (lane & 16) ? 8 : 0;
#pragma unroll
  for (int j = 0; j < 4; ++j) {
    const int hd = j * 16 + n;
    // inv_freq = 10000^(-(hd&~1)/64); ln(10000)/64 = 0.14391157
    const float freq = __expf(-(float)(hd & 62) * 0.14391157f);
    const bool odd = (hd & 1) != 0;
#pragma unroll
    for (int r = 0; r < 8; ++r) {
      const int srow = st * 16 + rbase + r;
      float sn, cs;
      __sincosf((float)srow * freq, &sn, &cs);
      const float vq = aq[j][r];
      const float oq = __shfl_xor(vq, 1, 32);
      const float rq = odd ? (oq * sn + vq * cs) : (vq * cs - oq * sn);
      const float vk = ak[j][r];
      const float ok = __shfl_xor(vk, 1, 32);
      const float rk = odd ? (ok * sn + vk * cs) : (vk * cs - ok * sn);
      qws[((size_t)h * S_LEN + srow) * HDIM + hd] = (__bf16)rq;
      kws[((size_t)h * S_LEN + srow) * HDIM + hd] = (__bf16)rk;
      vtws[((size_t)h * HDIM + hd) * S_LEN + srow] = (__bf16)av[j][r];
    }
  }
}

// ---------------- Kernel B: causal flash attention ----------------
// grid(S/16, H), block 32. One wave: 16 query rows, stream kv in tiles of 32.
__global__ __launch_bounds__(32)
void flash_attn_kernel(const __bf16* __restrict__ qws,
                       const __bf16* __restrict__ kws,
                       const __bf16* __restrict__ vtws,
                       __bf16* __restrict__ ctxws)   // [S][D] bf16
{
  __shared__ __align__(16) __bf16 pl[16 * 32];   // P tile staging (C->A relayout)
  const int lane = threadIdx.x;
  const int qt = blockIdx.x;
  const int h  = blockIdx.y;
  const int n = lane & 15;
  const int rbase = (lane & 16) ? 8 : 0;

  const __bf16* qrow = qws + ((size_t)h * S_LEN + (qt * 16 + n)) * HDIM;
  const v16bf aq0 = load_a_bf16(qrow + 0,  lane);
  const v16bf aq1 = load_a_bf16(qrow + 32, lane);

  v8f o0 = {}, o1 = {}, o2 = {}, o3 = {};
  float mrun[8], lrun[8];
#pragma unroll
  for (int r = 0; r < 8; ++r) { mrun[r] = -1e30f; lrun[r] = 0.0f; }

  const v8f zc = {};
  const __bf16* vbase0 = vtws + (size_t)h * HDIM * S_LEN;
  const int nkt = qt / 2 + 1;                 // kv tiles of 32 (causal)
  for (int kt = 0; kt < nkt; ++kt) {
    const int kvb = kt * 32;
    const __bf16* kr0 = kws + ((size_t)h * S_LEN + (kvb + n)) * HDIM;
    const __bf16* kr1 = kws + ((size_t)h * S_LEN + (kvb + 16 + n)) * HDIM;

    // speculative prefetch of next kv tile (-> global_prefetch_b8)
    if (kt + 1 < nkt) {
      __builtin_prefetch(kr0 + 32 * HDIM, 0, 0);
      __builtin_prefetch(kr1 + 32 * HDIM, 0, 0);
      __builtin_prefetch(vbase0 + (size_t)n * S_LEN + kvb + 32, 0, 0);
      __builtin_prefetch(vbase0 + (size_t)(48 + n) * S_LEN + kvb + 32, 0, 0);
    }

    v8f s0 = wmma_bf16(aq0, load_b_bf16(kr0 + 0,  lane), zc);
    s0     = wmma_bf16(aq1, load_b_bf16(kr0 + 32, lane), s0);
    v8f s1 = wmma_bf16(aq0, load_b_bf16(kr1 + 0,  lane), zc);
    s1     = wmma_bf16(aq1, load_b_bf16(kr1 + 32, lane), s1);

    // online softmax (each score row lives in one 16-lane half-wave)
#pragma unroll
    for (int r = 0; r < 8; ++r) {
      const int qg = qt * 16 + rbase + r;
      float v0 = s0[r] * 0.125f;              // 1/sqrt(HD), mup scale = 1
      float v1 = s1[r] * 0.125f;
      if (kvb + n > qg)      v0 = -1e30f;     // causal mask
      if (kvb + 16 + n > qg) v1 = -1e30f;
      float t = fmaxf(v0, v1);
      t = fmaxf(t, __shfl_xor(t, 1, 32));
      t = fmaxf(t, __shfl_xor(t, 2, 32));
      t = fmaxf(t, __shfl_xor(t, 4, 32));
      t = fmaxf(t, __shfl_xor(t, 8, 32));
      const float mnew = fmaxf(mrun[r], t);
      const float p0 = __expf(v0 - mnew);
      const float p1 = __expf(v1 - mnew);
      float rs = p0 + p1;
      rs += __shfl_xor(rs, 1, 32);
      rs += __shfl_xor(rs, 2, 32);
      rs += __shfl_xor(rs, 4, 32);
      rs += __shfl_xor(rs, 8, 32);
      const float sc = __expf(mrun[r] - mnew);
      lrun[r] = lrun[r] * sc + rs;
      mrun[r] = mnew;
      o0[r] *= sc; o1[r] *= sc; o2[r] *= sc; o3[r] *= sc;
      const int row = rbase + r;
      pl[row * 32 + n]      = (__bf16)p0;
      pl[row * 32 + 16 + n] = (__bf16)p1;
    }
    __syncthreads();

    // P (16x32) as A-fragment, V^T (kv-contiguous) as B-fragments
    const v16bf ap = load_a_bf16(&pl[(lane & 15) * 32], lane);
    const __bf16* vbase = vbase0 + kvb;
    o0 = wmma_bf16(ap, load_b_bf16(vbase + (size_t)(0 * 16 + n) * S_LEN, lane), o0);
    o1 = wmma_bf16(ap, load_b_bf16(vbase + (size_t)(1 * 16 + n) * S_LEN, lane), o1);
    o2 = wmma_bf16(ap, load_b_bf16(vbase + (size_t)(2 * 16 + n) * S_LEN, lane), o2);
    o3 = wmma_bf16(ap, load_b_bf16(vbase + (size_t)(3 * 16 + n) * S_LEN, lane), o3);
    __syncthreads();
  }

  // normalize + store ctx (bf16) in [S][D] layout for the output GEMM
#pragma unroll
  for (int r = 0; r < 8; ++r) {
    const float inv = 1.0f / lrun[r];
    const int srow = qt * 16 + rbase + r;
    __bf16* dst = ctxws + (size_t)srow * DMODEL + h * HDIM;
    dst[0 * 16 + n] = (__bf16)(o0[r] * inv);
    dst[1 * 16 + n] = (__bf16)(o1[r] * inv);
    dst[2 * 16 + n] = (__bf16)(o2[r] * inv);
    dst[3 * 16 + n] = (__bf16)(o3[r] * inv);
  }
}

// ---------------- Kernel C: output projection + bias ----------------
// grid(S/16, D/64), block 32: 4 N-tiles per wave sharing each A-fragment.
__global__ __launch_bounds__(32)
void out_proj_kernel(const __bf16* __restrict__ ctxws,
                     const __bf16* __restrict__ wob,
                     const float* __restrict__ bo,
                     float* __restrict__ out)
{
  const int lane = threadIdx.x;
  const int st = blockIdx.x;
  const int cb = blockIdx.y;            // 64-col block
  const int s_a  = st * 16 + (lane & 15);
  const int brow = lane & 15;

  v8f acc[4] = {};
  for (int kk = 0; kk < DMODEL; kk += 32) {
    const v16bf a = load_a_bf16(ctxws + (size_t)s_a * DMODEL + kk, lane);
#pragma unroll
    for (int j = 0; j < 4; ++j) {
      const size_t wrow = (size_t)(cb * 64 + j * 16 + brow) * DMODEL + kk;
      acc[j] = wmma_bf16(a, load_b_bf16(wob + wrow, lane), acc[j]);
    }
  }
  const int n = lane & 15;
  const int rbase = (lane & 16) ? 8 : 0;
#pragma unroll
  for (int j = 0; j < 4; ++j) {
    const int oc = cb * 64 + j * 16 + n;
    const float bias = bo[oc];
#pragma unroll
    for (int r = 0; r < 8; ++r)
      out[(size_t)(st * 16 + rbase + r) * DMODEL + oc] = acc[j][r] + bias;
  }
}

extern "C" void kernel_launch(void* const* d_in, const int* in_sizes, int n_in,
                              void* d_out, int out_size, void* d_ws, size_t ws_size,
                              hipStream_t stream) {
  (void)in_sizes; (void)n_in; (void)out_size; (void)ws_size;
  const float* x  = (const float*)d_in[0];
  const float* Wq = (const float*)d_in[1];
  const float* Wk = (const float*)d_in[2];
  const float* Wv = (const float*)d_in[3];
  const float* Wo = (const float*)d_in[4];
  const float* bo = (const float*)d_in[5];
  float* out = (float*)d_out;

  const size_t XN  = (size_t)S_LEN * DMODEL;        // 4M elems
  const size_t WN  = (size_t)DMODEL * DMODEL;       // 1M elems
  const size_t HSD = (size_t)NHEAD * S_LEN * HDIM;  // 4M elems
  __bf16* xb    = (__bf16*)d_ws;
  __bf16* wqb   = xb   + XN;
  __bf16* wkb   = wqb  + WN;
  __bf16* wvb   = wkb  + WN;
  __bf16* wob   = wvb  + WN;
  __bf16* qws   = wob  + WN;
  __bf16* kws   = qws  + HSD;
  __bf16* vtws  = kws  + HSD;
  __bf16* ctxws = vtws + HSD;                       // total 24M elems = 48 MB

  // one-shot bf16 conversion of activations + weights
  cvt_bf16_kernel<<<dim3((int)(XN / 8 / 256)), dim3(256), 0, stream>>>(x,  xb,  (int)(XN / 8));
  cvt_bf16_kernel<<<dim3((int)(WN / 8 / 256)), dim3(256), 0, stream>>>(Wq, wqb, (int)(WN / 8));
  cvt_bf16_kernel<<<dim3((int)(WN / 8 / 256)), dim3(256), 0, stream>>>(Wk, wkb, (int)(WN / 8));
  cvt_bf16_kernel<<<dim3((int)(WN / 8 / 256)), dim3(256), 0, stream>>>(Wv, wvb, (int)(WN / 8));
  cvt_bf16_kernel<<<dim3((int)(WN / 8 / 256)), dim3(256), 0, stream>>>(Wo, wob, (int)(WN / 8));

  const dim3 blk(32, 1, 1);
  qkv_rope_kernel<<<dim3(S_LEN / 16, NHEAD), blk, 0, stream>>>(
      xb, wqb, wkb, wvb, qws, kws, vtws);
  flash_attn_kernel<<<dim3(S_LEN / 16, NHEAD), blk, 0, stream>>>(
      qws, kws, vtws, ctxws);
  out_proj_kernel<<<dim3(S_LEN / 16, DMODEL / 64), blk, 0, stream>>>(
      ctxws, wob, bo, out);
}
